// KomplexNet_44727789421110
// MI455X (gfx1250) — matrix operand order repaired and checked by reading, compiled, not run on hardware
//
#include <hip/hip_runtime.h>
#include <math.h>

typedef __attribute__((ext_vector_type(16))) _Float16 v16h;
typedef __attribute__((ext_vector_type(8)))  _Float16 v8h;
typedef __attribute__((ext_vector_type(8)))  float    v8f;

#define B_N   32
#define F_N   32
#define HW_N  32
#define CK_N  8
#define G_N   5
#define DIM_N 32
#define TSTEPS 8
#define PI_F  3.14159265358979f
#define EPS_K 0.1f
#define LR_KU 0.1f

static __device__ __forceinline__ float blockReduceSum(float v, float* sh) {
  int t = threadIdx.x;
  sh[t] = v; __syncthreads();
  for (int s = blockDim.x >> 1; s > 0; s >>= 1) {
    if (t < s) sh[t] += sh[t + s];
    __syncthreads();
  }
  float r = sh[0]; __syncthreads();
  return r;
}

static __device__ __forceinline__ v8h zero_v8h() {
  v8h z = {(_Float16)0, (_Float16)0, (_Float16)0, (_Float16)0,
           (_Float16)0, (_Float16)0, (_Float16)0, (_Float16)0};
  return z;
}

// ---------------- front-end: 3->8 channel 2D conv + relu, plus tanh ----------------
__global__ void k_down(const float* __restrict__ in, const float* __restrict__ wd,
                       float* __restrict__ amps, float* __restrict__ tb) {
  int idx = blockIdx.x * blockDim.x + threadIdx.x;     // B*F*CK*1024
  int x = idx & 31, y = (idx >> 5) & 31, c = (idx >> 10) & 7;
  int img = idx >> 13;           // b*F + f
  int f = img & 31, b = img >> 5;
  float acc = 0.f;
  for (int ci = 0; ci < 3; ci++) {
    const float* ip = in + (((size_t)(b * 3 + ci) * F_N + f) << 10);
    for (int ky = 0; ky < 3; ky++) {
      int iy = y + ky - 1;
      if ((unsigned)iy >= 32u) continue;
      for (int kx = 0; kx < 3; kx++) {
        int ix = x + kx - 1;
        if ((unsigned)ix >= 32u) continue;
        acc += ip[iy * 32 + ix] * wd[((c * 3 + ci) * 3 + ky) * 3 + kx];
      }
    }
  }
  acc = fmaxf(acc, 0.f);
  amps[idx] = acc;
  tb[idx]   = tanhf(acc);
}

__global__ void k_init_phase(const float* __restrict__ p0, float* __restrict__ p) {
  int i = blockIdx.x * blockDim.x + threadIdx.x;
  p[i] = p0[i] * (2.f * PI_F) - PI_F;
}

// ---------------- Kuramoto update ----------------
__global__ void k_u1(const float* __restrict__ p, const float* __restrict__ tb, int f,
                     float* __restrict__ cp, float* __restrict__ sp,
                     float* __restrict__ bc, float* __restrict__ bs) {
  int idx = blockIdx.x * blockDim.x + threadIdx.x;   // B*CK*1024
  int b = idx >> 13, c = (idx >> 10) & 7, pos = idx & 1023;
  float pv = p[idx];
  float cv = cosf(pv), sv = sinf(pv);
  float tv = tb[((((size_t)b * F_N + f) * CK_N + c) << 10) + pos];
  cp[idx] = cv; sp[idx] = sv; bc[idx] = cv * tv; bs[idx] = sv * tv;
}

__global__ void k_usum(const float* __restrict__ bc, const float* __restrict__ bs,
                       float* __restrict__ Sc, float* __restrict__ Ss) {
  __shared__ float sh[256];
  int b = blockIdx.x;
  float ac = 0.f, as_ = 0.f;
  for (int i = threadIdx.x; i < 8192; i += 256) {
    ac  += bc[b * 8192 + i];
    as_ += bs[b * 8192 + i];
  }
  float rc = blockReduceSum(ac, sh);
  float rs = blockReduceSum(as_, sh);
  if (threadIdx.x == 0) { Sc[b] = rc; Ss[b] = rs; }
}

__global__ void k_u2(const float* __restrict__ kern, const float* __restrict__ bc,
                     const float* __restrict__ bs, const float* __restrict__ cp,
                     const float* __restrict__ sp, const float* __restrict__ Sc,
                     const float* __restrict__ Ss, float* __restrict__ p) {
  __shared__ float sk[CK_N * CK_N * 25];
  for (int i = threadIdx.x; i < CK_N * CK_N * 25; i += blockDim.x) sk[i] = kern[i];
  __syncthreads();
  int idx = blockIdx.x * blockDim.x + threadIdx.x;
  int b = idx >> 13, co = (idx >> 10) & 7, y = (idx >> 5) & 31, x = idx & 31;
  float Cc = 0.f, Cs = 0.f;
  for (int ci = 0; ci < 8; ci++) {
    const float* bcp = bc + (((size_t)b * 8 + ci) << 10);
    const float* bsp = bs + (((size_t)b * 8 + ci) << 10);
    const float* kp  = sk + (co * 8 + ci) * 25;
    for (int ky = 0; ky < 5; ky++) {
      int iy = y + ky - 2;
      if ((unsigned)iy >= 32u) continue;
      for (int kx = 0; kx < 5; kx++) {
        int ix = x + kx - 2;
        if ((unsigned)ix >= 32u) continue;
        float w = kp[ky * 5 + kx];
        Cc += w * bcp[iy * 32 + ix];
        Cs += w * bsp[iy * 32 + ix];
      }
    }
  }
  p[idx] += LR_KU * (cp[idx] * (Cs - EPS_K * Ss[b]) - sp[idx] * (Cc - EPS_K * Sc[b]));
}

__global__ void k_copy(const float* __restrict__ s, float* __restrict__ d) {
  int i = blockIdx.x * blockDim.x + threadIdx.x;
  d[i] = s[i];
}

// ---------------- synchrony loss ----------------
__global__ void k_s1(const float* __restrict__ p, const int* __restrict__ masks, int f,
                     float* __restrict__ xx, float* __restrict__ yy, float* __restrict__ gs) {
  __shared__ float sh[256];
  int bg = blockIdx.x;          // b*4 + g
  int b = bg >> 2, g = bg & 3;
  const int* mp = masks + (((size_t)(b * G_N + (g + 1)) * F_N + f) << 10);
  float accc[8], accs[8];
  for (int c = 0; c < 8; c++) { accc[c] = 0.f; accs[c] = 0.f; }
  float cnt = 0.f;
  for (int pix = threadIdx.x; pix < 1024; pix += 256) {
    int mv = mp[pix];
    float mvf = (float)mv;
    cnt += mvf;
    if (mv > 0) {
      for (int c = 0; c < 8; c++) {
        float pv = p[(((size_t)b * 8 + c) << 10) + pix];
        float ph = atan2f(sinf(pv), cosf(pv));
        float a = ph * mvf;
        accc[c] += cosf(a);
        accs[c] += sinf(a);
      }
    }
  }
  for (int c = 0; c < 8; c++) {
    float r = blockReduceSum(accc[c], sh);
    if (threadIdx.x == 0) xx[bg * 8 + c] = r;
  }
  for (int c = 0; c < 8; c++) {
    float r = blockReduceSum(accs[c], sh);
    if (threadIdx.x == 0) yy[bg * 8 + c] = r;
  }
  float rc = blockReduceSum(cnt, sh);
  if (threadIdx.x == 0) gs[bg] = rc;
}

__global__ void k_s2(const float* __restrict__ xx, const float* __restrict__ yy,
                     const float* __restrict__ gs, float* __restrict__ loss) {
  __shared__ float sh[32];
  int b = threadIdx.x;           // 32 threads
  float synch_acc = 0.f;
  float ang[4];
  for (int g = 0; g < 4; g++) {
    int bg = b * 4 + g;
    float gsv = gs[bg];
    if (gsv == 0.f) gsv = 1.f;
    float sxx = 0.f, syy = 0.f, go = 0.f;
    for (int c = 0; c < 8; c++) {
      float xv = xx[bg * 8 + c], yv = yy[bg * 8 + c];
      go += sqrtf(xv * xv + yv * yv) / gsv;
      sxx += xv; syy += yv;
    }
    synch_acc += go / 8.f;
    ang[g] = atan2f(syy / 8.f, sxx / 8.f);
  }
  float synch = 1.f - synch_acc / 4.f;
  float desynch = 0.f;
  for (int k = 1; k <= 2; k++) {
    float cs = 0.f, sn = 0.f;
    for (int g = 0; g < 4; g++) { cs += cosf((float)k * ang[g]); sn += sinf((float)k * ang[g]); }
    desynch += (cs * cs + sn * sn) / (2.f * 4.f * (float)(k * k));
  }
  float val = (synch + desynch) / 3.5f;   // lb = 1 + 0.5*4*(1 + 1/4)
  sh[b] = val; __syncthreads();
  if (b == 0) {
    float s = 0.f;
    for (int i = 0; i < 32; i++) s += sh[i];
    atomicAdd(loss, s / (float)(B_N * F_N));
  }
}

// ---------------- complex feature build: f16 channel-last [b][f][h][w][c] ----------------
__global__ void k_zrzi(const float* __restrict__ amps, const float* __restrict__ phall,
                       _Float16* __restrict__ xhr, _Float16* __restrict__ xhi) {
  int idx = blockIdx.x * blockDim.x + threadIdx.x;   // (b, c, f, h, w)
  int pos = idx & 1023;
  int f = (idx >> 10) & 31;
  int c = (idx >> 15) & 7;
  int b = idx >> 18;
  int h = pos >> 5, w = pos & 31;
  float a  = amps [((((size_t)(b * F_N + f) * CK_N) + c) << 10) + pos];
  float ph = phall[((((size_t)(f * B_N + b) * CK_N) + c) << 10) + pos];
  size_t off = ((((size_t)b * 32 + f) * 32 + h) * 32 + w) * 8 + c;   // NDHWC
  xhr[off] = (_Float16)(a * cosf(ph));
  xhi[off] = (_Float16)(a * sinf(ph));
}

// ---------------- weight repack: f32 (Cout,Cin,27) -> f16 tap-major (Cout, Kpad) ----------------
// k = tap*Cin + cin ; zero-padded to Kpad (multiple of 32)
__global__ void k_w2h(const float* __restrict__ w, _Float16* __restrict__ wh,
                      int Cin, int lc, int Kpad, int n) {
  int idx = blockIdx.x * blockDim.x + threadIdx.x;
  if (idx >= n) return;
  int co = idx / Kpad, k = idx - co * Kpad;
  int Ktot = Cin * 27;
  float v = 0.f;
  if (k < Ktot) {
    int r = k >> lc;
    int cin = k & (Cin - 1);
    v = w[((size_t)co * Cin + cin) * 27 + r];
  }
  wh[idx] = (_Float16)v;
}

// ---------------- implicit-GEMM conv3d via V_WMMA_F32_16X16X32_F16 ----------------
// stride (1,2,2), pad (1,1,1). One wave -> 16(out-ch) x 64(voxel) tile; 4 WMMAs share one
// A fragment per K-step. K-order = tap*Cin+cin so every fragment is aligned b128 loads.
__global__ void k_conv3d_wmma(const _Float16* __restrict__ xh,   // [B][D][Hin][Win][Cin] f16
                              const _Float16* __restrict__ wh,   // [Cout][Kpad] f16
                              const float* __restrict__ bias, float* __restrict__ out,
                              int Cin, int lc, int Cout, int D, int Hin, int Win,
                              int Hout, int Wout, int Kpad) {
  const int lane = threadIdx.x;        // 32 threads = 1 wave, EXEC all 1s at WMMA
  const int lm = lane & 15;
  const int hi = lane >> 4;
  const int cout0 = blockIdx.y * 16;
  const int b = blockIdx.z;
  const int HWo = Hout * Wout;
  const int Ktot = Cin * 27;

  int od[4], oh[4], ow[4];
  for (int t = 0; t < 4; t++) {
    int ng = blockIdx.x * 64 + t * 16 + lm;
    od[t] = ng / HWo;
    int rem = ng - od[t] * HWo;
    oh[t] = rem / Wout;
    ow[t] = rem - oh[t] * Wout;
  }

  const _Float16* wrow = wh + (size_t)(cout0 + lm) * Kpad;
  const _Float16* xb = xh + (size_t)b * D * Hin * Win * Cin;

  union Acc { v8f v; float e[8]; } acc[4];
  acc[0].v = (v8f){0.f, 0.f, 0.f, 0.f, 0.f, 0.f, 0.f, 0.f};
  acc[1].v = acc[0].v; acc[2].v = acc[0].v; acc[3].v = acc[0].v;

  for (int kk0 = 0; kk0 < Ktot; kk0 += 32) {
    union Frag { v16h v; struct { v8h lo; v8h hi; } h; } af, bf[4];
    // A fragment: lanes>=16 hold K+16; j0..7 -> [kk0+16hi, +8), j8..15 -> [kk0+16hi+8, +8)
    af.h.lo = *(const v8h*)(wrow + kk0 + 16 * hi);
    af.h.hi = *(const v8h*)(wrow + kk0 + 16 * hi + 8);
    // B fragments: lanes>=16 hold K+16; contiguous 16 K split into two aligned 8-chunks
    int ks = kk0 + 16 * hi;
    for (int cnk = 0; cnk < 2; cnk++) {
      int kc = ks + 8 * cnk;
      int r = kc >> lc;                 // tap index (Cin is power of two)
      int cin = kc & (Cin - 1);
      v8h ch[4];
      ch[0] = zero_v8h(); ch[1] = ch[0]; ch[2] = ch[0]; ch[3] = ch[0];
      if (r < 27) {
        int kd = r / 9, r2 = r - kd * 9, kh = r2 / 3, kw = r2 - kh * 3;
        for (int t = 0; t < 4; t++) {
          int id = od[t] + kd - 1, ih = oh[t] * 2 + kh - 1, iw = ow[t] * 2 + kw - 1;
          if ((unsigned)id < (unsigned)D && (unsigned)ih < (unsigned)Hin &&
              (unsigned)iw < (unsigned)Win)
            ch[t] = *(const v8h*)(xb + (((size_t)id * Hin + ih) * Win + iw) * Cin + cin);
        }
      }
      if (cnk == 0) { for (int t = 0; t < 4; t++) bf[t].h.lo = ch[t]; }
      else          { for (int t = 0; t < 4; t++) bf[t].h.hi = ch[t]; }
    }
    acc[0].v = __builtin_amdgcn_wmma_f32_16x16x32_f16(
        false, af.v, false, bf[0].v, (short)0, acc[0].v, false, false);
    acc[1].v = __builtin_amdgcn_wmma_f32_16x16x32_f16(
        false, af.v, false, bf[1].v, (short)0, acc[1].v, false, false);
    acc[2].v = __builtin_amdgcn_wmma_f32_16x16x32_f16(
        false, af.v, false, bf[2].v, (short)0, acc[2].v, false, false);
    acc[3].v = __builtin_amdgcn_wmma_f32_16x16x32_f16(
        false, af.v, false, bf[3].v, (short)0, acc[3].v, false, false);
  }

  // C/D layout: VGPR r -> (M=r, N=lane) lanes 0-15; (M=r+8, N=lane-16) lanes 16-31
  for (int r = 0; r < 8; r++) {
    int co = cout0 + r + hi * 8;
    float bv = bias[co];
    for (int t = 0; t < 4; t++) {
      out[(((size_t)(b * Cout + co) * D + od[t]) * Hout + oh[t]) * Wout + ow[t]] =
          acc[t].e[r] + bv;
    }
  }
}

// ---------------- magnitude batchnorm + relu (phase preserved) ----------------
__global__ void k_bn_zero(float* s, float* q, int C) {
  int i = threadIdx.x;
  if (i < C) { s[i] = 0.f; q[i] = 0.f; }
}

__global__ void k_bn_stat(const float* __restrict__ zr, const float* __restrict__ zi,
                          float* __restrict__ s, float* __restrict__ q, int C, int P) {
  __shared__ float sh[256];
  int bc = blockIdx.x;            // b*C + c
  int c = bc % C;
  const float* pr = zr + (size_t)bc * P;
  const float* pi = zi + (size_t)bc * P;
  float a = 0.f, a2 = 0.f;
  for (int i = threadIdx.x; i < P; i += 256) {
    float m = sqrtf(pr[i] * pr[i] + pi[i] * pi[i] + 1e-12f);
    a += m; a2 += m * m;
  }
  float r  = blockReduceSum(a, sh);
  float r2 = blockReduceSum(a2, sh);
  if (threadIdx.x == 0) { atomicAdd(s + c, r); atomicAdd(q + c, r2); }
}

// applies BN+relu on magnitude in-place (f32 NCDHW) and emits f16 NDHWC for the next conv
__global__ void k_bn_apply(float* __restrict__ zr, float* __restrict__ zi,
                           const float* __restrict__ s, const float* __restrict__ q,
                           const float* __restrict__ g, const float* __restrict__ be,
                           int C, int D, int H, int W,
                           _Float16* __restrict__ xhr, _Float16* __restrict__ xhi) {
  int idx = blockIdx.x * blockDim.x + threadIdx.x;
  int P = D * H * W;
  int pp = idx % P;
  int c = (idx / P) % C;
  int b = idx / (P * C);
  float n = (float)B_N * (float)P;
  float mu = s[c] / n;
  float var = q[c] / n - mu * mu;
  float xr = zr[idx], xi = zi[idx];
  float mag = sqrtf(xr * xr + xi * xi + 1e-12f);
  float xn = (mag - mu) * rsqrtf(var + 1e-3f);
  float m = fmaxf(xn * g[c] + be[c], 0.f);
  float sc = m / mag;
  float vr = xr * sc, vi = xi * sc;
  zr[idx] = vr;
  zi[idx] = vi;
  if (xhr) {
    int d = pp / (H * W);
    int rem = pp - d * (H * W);
    int h = rem / W, w = rem - h * W;
    size_t off = ((((size_t)b * D + d) * H + h) * W + w) * C + c;   // NDHWC
    xhr[off] = (_Float16)vr;
    xhi[off] = (_Float16)vi;
  }
}

// ---------------- final projection + magnitude ----------------
__global__ void k_final(const float* __restrict__ zr, const float* __restrict__ zi,
                        const float* __restrict__ wout, const float* __restrict__ br,
                        const float* __restrict__ bi, float* __restrict__ out) {
  __shared__ float sh[256];
  int b = blockIdx.x;
  const float* fr = zr + (size_t)b * 32768;
  const float* fi = zi + (size_t)b * 32768;
  float ar = 0.f, ai = 0.f;
  for (int i = threadIdx.x; i < 32768; i += 256) {
    float w = wout[i];
    ar += fr[i] * w;
    ai += fi[i] * w;
  }
  float rr = blockReduceSum(ar, sh);
  float ri = blockReduceSum(ai, sh);
  if (threadIdx.x == 0) {
    float orr = rr + br[0], oii = ri + bi[0];
    out[b] = sqrtf(orr * orr + oii * oii + 1e-12f);
  }
}

__global__ void k_zero_loss(float* l) { *l = 0.f; }

// ---------------- host orchestration ----------------
extern "C" void kernel_launch(void* const* d_in, const int* in_sizes, int n_in,
                              void* d_out, int out_size, void* d_ws, size_t ws_size,
                              hipStream_t stream) {
  (void)in_sizes; (void)n_in; (void)out_size; (void)ws_size;
  const float* input   = (const float*)d_in[0];
  const int*   masks   = (const int*)  d_in[1];
  const float* phases0 = (const float*)d_in[2];
  const float* w_down  = (const float*)d_in[3];
  const float* kern    = (const float*)d_in[4];
  const float* w_pre   = (const float*)d_in[5];
  const float* b_pre_r = (const float*)d_in[6];
  const float* b_pre_i = (const float*)d_in[7];
  const float* g1      = (const float*)d_in[8];
  const float* be1     = (const float*)d_in[9];
  const float* w_c1    = (const float*)d_in[10];
  const float* b_c1_r  = (const float*)d_in[11];
  const float* b_c1_i  = (const float*)d_in[12];
  const float* g2      = (const float*)d_in[13];
  const float* be2     = (const float*)d_in[14];
  const float* w_c2    = (const float*)d_in[15];
  const float* b_c2_r  = (const float*)d_in[16];
  const float* b_c2_i  = (const float*)d_in[17];
  const float* g3      = (const float*)d_in[18];
  const float* be3     = (const float*)d_in[19];
  const float* w_out   = (const float*)d_in[20];
  const float* b_out_r = (const float*)d_in[21];
  const float* b_out_i = (const float*)d_in[22];
  float* out = (float*)d_out;

  // ---- workspace layout (byte offsets, 256B aligned) ----
  char* wsb = (char*)d_ws;
  size_t o = 0;
  auto alloc = [&](size_t bytes) -> void* {
    void* ptr = wsb + o;
    o += (bytes + 255) & ~(size_t)255;
    return ptr;
  };
  const size_t SZ_AMPS = (size_t)B_N * F_N * CK_N * 1024;  // 8,388,608 elems
  const size_t SZ_P    = (size_t)B_N * CK_N * 1024;        // 262,144 elems
  float* amps  = (float*)alloc(SZ_AMPS * 4);
  float* tbuf  = (float*)alloc(SZ_AMPS * 4);
  float* phall = (float*)alloc(SZ_AMPS * 4);
  float* p     = (float*)alloc(SZ_P * 4);
  float* cp    = (float*)alloc(SZ_P * 4);
  float* sp    = (float*)alloc(SZ_P * 4);
  float* bc    = (float*)alloc(SZ_P * 4);
  float* bs    = (float*)alloc(SZ_P * 4);
  float* Sc    = (float*)alloc(32 * 4);
  float* Ss    = (float*)alloc(32 * 4);
  float* xx    = (float*)alloc(1024 * 4);
  float* yy    = (float*)alloc(1024 * 4);
  float* gs    = (float*)alloc(128 * 4);
  float* bns   = (float*)alloc(64 * 4);
  float* bnq   = (float*)alloc(64 * 4);
  const int KP1 = 224, KP2 = 864, KP3 = 1728;              // Kpad (multiples of 32)
  _Float16* whpre = (_Float16*)alloc((size_t)32 * KP1 * 2);
  _Float16* wh1   = (_Float16*)alloc((size_t)64 * KP2 * 2);
  _Float16* wh2   = (_Float16*)alloc((size_t)64 * KP3 * 2);
  _Float16* x1hr  = (_Float16*)alloc(SZ_AMPS * 2);         // layer1 in, NDHWC f16
  _Float16* x1hi  = (_Float16*)alloc(SZ_AMPS * 2);
  // aliased regions (sequential lifetime):
  float* o1r = amps;                                       // layer1 out f32 (B,32,32,16,16)
  float* o1i = tbuf;
  _Float16* x2hr = (_Float16*)phall;                       // layer2 in f16 (16 MB each)
  _Float16* x2hi = (_Float16*)((char*)phall + 16777216);
  float* o2r = (float*)x1hr;                               // layer2 out f32 (16 MB each)
  float* o2i = (float*)x1hi;
  _Float16* x3hr = (_Float16*)amps;                        // layer3 in f16 (8.4 MB each)
  _Float16* x3hi = (_Float16*)tbuf;
  float* o3r = phall;                                      // layer3 out f32 (4.2 MB each)
  float* o3i = (float*)((char*)phall + 4194304);

  dim3 blk(256);
  k_zero_loss<<<1, 1, 0, stream>>>(out + 32);
  k_down<<<dim3((unsigned)(SZ_AMPS / 256)), blk, 0, stream>>>(input, w_down, amps, tbuf);
  k_init_phase<<<dim3((unsigned)(SZ_P / 256)), blk, 0, stream>>>(phases0, p);
  // repack weights to f16 tap-major (once per call)
  k_w2h<<<dim3((32 * KP1 + 255) / 256), blk, 0, stream>>>(w_pre, whpre, 8, 3, KP1, 32 * KP1);
  k_w2h<<<dim3((64 * KP2 + 255) / 256), blk, 0, stream>>>(w_c1, wh1, 32, 5, KP2, 64 * KP2);
  k_w2h<<<dim3((64 * KP3 + 255) / 256), blk, 0, stream>>>(w_c2, wh2, 64, 6, KP3, 64 * KP3);

  auto kstep = [&](int f) {
    k_u1<<<dim3((unsigned)(SZ_P / 256)), blk, 0, stream>>>(p, tbuf, f, cp, sp, bc, bs);
    k_usum<<<dim3(32), blk, 0, stream>>>(bc, bs, Sc, Ss);
    k_u2<<<dim3((unsigned)(SZ_P / 256)), blk, 0, stream>>>(kern, bc, bs, cp, sp, Sc, Ss, p);
  };

  for (int t = 0; t < TSTEPS; t++) kstep(0);           // frame 0: 8 timesteps
  k_copy<<<dim3((unsigned)(SZ_P / 256)), blk, 0, stream>>>(p, phall);
  k_s1<<<dim3(128), blk, 0, stream>>>(p, masks, 0, xx, yy, gs);
  k_s2<<<dim3(1), dim3(32), 0, stream>>>(xx, yy, gs, out + 32);
  for (int f = 1; f < F_N; f++) {                      // frames 1..31: one step each
    kstep(f);
    k_copy<<<dim3((unsigned)(SZ_P / 256)), blk, 0, stream>>>(p, phall + (size_t)f * SZ_P);
    k_s1<<<dim3(128), blk, 0, stream>>>(p, masks, f, xx, yy, gs);
    k_s2<<<dim3(1), dim3(32), 0, stream>>>(xx, yy, gs, out + 32);
  }

  k_zrzi<<<dim3((unsigned)(SZ_AMPS / 256)), blk, 0, stream>>>(amps, phall, x1hr, x1hi);

  dim3 wave(32);
  // layer 1: 8 -> 32 ch, 32x32x32 -> 32x16x16 ; N=8192 -> 128 waves of 64 cols
  k_conv3d_wmma<<<dim3(128, 2, 32), wave, 0, stream>>>(x1hr, whpre, b_pre_r, o1r, 8, 3, 32, 32, 32, 32, 16, 16, KP1);
  k_conv3d_wmma<<<dim3(128, 2, 32), wave, 0, stream>>>(x1hi, whpre, b_pre_i, o1i, 8, 3, 32, 32, 32, 32, 16, 16, KP1);
  k_bn_zero<<<1, 64, 0, stream>>>(bns, bnq, 32);
  k_bn_stat<<<dim3(32 * 32), blk, 0, stream>>>(o1r, o1i, bns, bnq, 32, 8192);
  k_bn_apply<<<dim3(8388608 / 256), blk, 0, stream>>>(o1r, o1i, bns, bnq, g1, be1, 32, 32, 16, 16, x2hr, x2hi);

  // layer 2: 32 -> 64 ch, 32x16x16 -> 32x8x8 ; N=2048 -> 32 waves
  k_conv3d_wmma<<<dim3(32, 4, 32), wave, 0, stream>>>(x2hr, wh1, b_c1_r, o2r, 32, 5, 64, 32, 16, 16, 8, 8, KP2);
  k_conv3d_wmma<<<dim3(32, 4, 32), wave, 0, stream>>>(x2hi, wh1, b_c1_i, o2i, 32, 5, 64, 32, 16, 16, 8, 8, KP2);
  k_bn_zero<<<1, 64, 0, stream>>>(bns, bnq, 64);
  k_bn_stat<<<dim3(32 * 64), blk, 0, stream>>>(o2r, o2i, bns, bnq, 64, 2048);
  k_bn_apply<<<dim3(4194304 / 256), blk, 0, stream>>>(o2r, o2i, bns, bnq, g2, be2, 64, 32, 8, 8, x3hr, x3hi);

  // layer 3: 64 -> 64 ch, 32x8x8 -> 32x4x4 ; N=512 -> 8 waves
  k_conv3d_wmma<<<dim3(8, 4, 32), wave, 0, stream>>>(x3hr, wh2, b_c2_r, o3r, 64, 6, 64, 32, 8, 8, 4, 4, KP3);
  k_conv3d_wmma<<<dim3(8, 4, 32), wave, 0, stream>>>(x3hi, wh2, b_c2_i, o3i, 64, 6, 64, 32, 8, 8, 4, 4, KP3);
  k_bn_zero<<<1, 64, 0, stream>>>(bns, bnq, 64);
  k_bn_stat<<<dim3(32 * 64), blk, 0, stream>>>(o3r, o3i, bns, bnq, 64, 512);
  k_bn_apply<<<dim3(1048576 / 256), blk, 0, stream>>>(o3r, o3i, bns, bnq, g3, be3, 64, 32, 4, 4,
                                                      (_Float16*)nullptr, (_Float16*)nullptr);

  k_final<<<dim3(32), blk, 0, stream>>>(o3r, o3i, w_out, b_out_r, b_out_i, out);
}
